// Transformer_41463614276150
// MI455X (gfx1250) — compile-verified, matrix operand order checked by reading, not run on hardware
//
#include <hip/hip_runtime.h>

#define B_      8
#define SEQ     256
#define HID     768
#define NHEADS  8
#define DKH     96
#define FFD     3072
#define VOC     16384
#define NLAYERS 6
#define KTOP    13
#define NRAND   26
#define MOD_SCALE 1.2f
#define PAD_TOK 1

// GEMM workgroup tiling
#define BM 32
#define BN 64
#define BK 32

typedef __attribute__((ext_vector_type(16))) __bf16 v16bf;
typedef __attribute__((ext_vector_type(2)))  __bf16 v2bf;
typedef __attribute__((ext_vector_type(8)))  float  v8f;

// ---------- TDM availability ----------
#if defined(__gfx1250__) && __has_builtin(__builtin_amdgcn_tensor_load_to_lds)
#define HAVE_TDM 1
#if __has_include(<hip/amd_detail/amd_gfx1250_TDM.h>)
#define TDM_ARGS6 1   // therock-10 headers -> 6-arg builtin
#else
#define TDM_ARGS6 0   // ROCm 7.2 -> 5-arg builtin (probe-verified types)
#endif
#else
#define HAVE_TDM 0
#endif

// ---------- helpers ----------

// Backend selects native v_cvt_pk_bf16_f32 (confirmed in round-2 disasm).
__device__ __forceinline__ v2bf pk_bf16(float a, float b) {
  v2bf r; r[0] = (__bf16)a; r[1] = (__bf16)b; return r;
}

__device__ __forceinline__ v8f wmma_bf16(v16bf a, v16bf b, v8f c) {
  return __builtin_amdgcn_wmma_f32_16x16x32_bf16(false, a, false, b, (short)0, c,
                                                 false, false);
}

__device__ __forceinline__ float wave_sum(float v) {
  #pragma unroll
  for (int m = 16; m > 0; m >>= 1) v += __shfl_xor(v, m, 32);
  return v;
}
__device__ __forceinline__ float wave_max(float v) {
  #pragma unroll
  for (int m = 16; m > 0; m >>= 1) v = fmaxf(v, __shfl_xor(v, m, 32));
  return v;
}

// A-fragment (16x32 bf16, MxK) from row-major fp32 (global or LDS).
__device__ __forceinline__ v16bf load_frag_row(const float* rowbase, int k0, int half) {
  const float4* p0 = (const float4*)(rowbase + k0 + half * 8);
  const float4* p1 = (const float4*)(rowbase + k0 + 16 + half * 8);
  float4 a = p0[0], b = p0[1], c = p1[0], d = p1[1];
  union { v16bf v; v2bf p[8]; } f;
  f.p[0] = pk_bf16(a.x, a.y); f.p[1] = pk_bf16(a.z, a.w);
  f.p[2] = pk_bf16(b.x, b.y); f.p[3] = pk_bf16(b.z, b.w);
  f.p[4] = pk_bf16(c.x, c.y); f.p[5] = pk_bf16(c.z, c.w);
  f.p[6] = pk_bf16(d.x, d.y); f.p[7] = pk_bf16(d.z, d.w);
  return f.v;
}

// B-fragment (32x16 bf16, KxN) from fp32 with column n fixed, row stride ld.
__device__ __forceinline__ v16bf load_frag_col(const float* colbase, int ld, int k0, int half) {
  union { v16bf v; v2bf p[8]; } f;
  #pragma unroll
  for (int e = 0; e < 16; e += 2) {
    int k = k0 + (e < 8 ? half * 8 + e : 16 + half * 8 + (e - 8));
    float x0 = colbase[(size_t)k * ld];
    float x1 = colbase[(size_t)(k + 1) * ld];
    f.p[e >> 1] = pk_bf16(x0, x1);
  }
  return f.v;
}

#if HAVE_TDM
typedef __attribute__((ext_vector_type(4))) unsigned int u32x4;
typedef __attribute__((ext_vector_type(8))) int i32x8;
typedef __attribute__((ext_vector_type(4))) int i32x4;

// TDM: DMA a rows x cols fp32 2D tile (row stride ld elements, tensor rows tdim1)
// from global memory into contiguous LDS at lds_dst.  D# layout per CDNA5 ISA ch.8.
__device__ __forceinline__ void tdm_load_tile_2d(const float* gsrc, void* lds_dst,
                                                 int rows, int cols, int ld, int tdim1) {
  unsigned lds_addr = (unsigned)(size_t)lds_dst;       // flat addr[31:0] == LDS offset
  unsigned long long ga = (unsigned long long)(size_t)gsrc;
  u32x4 g0;
  g0[0] = 1u;                                          // count=1, user descriptor
  g0[1] = lds_addr;                                    // lds_addr [63:32]
  g0[2] = (unsigned)(ga & 0xffffffffu);                // global_addr low
  g0[3] = (unsigned)((ga >> 32) & 0x01ffffffu) | (2u << 30);  // addr[56:32] | type=2
  i32x8 g1;
  g1[0] = (int)(2u << 16);                             // data_size=4B, no multicast
  g1[1] = (int)(((unsigned)ld & 0xffffu) << 16);       // tensor_dim0[15:0] @ bit48
  g1[2] = (int)((((unsigned)ld >> 16) & 0xffffu) |     // tensor_dim0[31:16]
                (((unsigned)tdim1 & 0xffffu) << 16));  // tensor_dim1[15:0]
  g1[3] = (int)((((unsigned)tdim1 >> 16) & 0xffffu) |  // tensor_dim1[31:16]
                (((unsigned)cols & 0xffffu) << 16));   // tile_dim0
  g1[4] = rows & 0xffff;                               // tile_dim1 (tile_dim2=0)
  g1[5] = ld;                                          // tensor_dim0_stride[31:0]
  g1[6] = 0;
  g1[7] = 0;
  i32x4 z = {0, 0, 0, 0};
#if TDM_ARGS6
  i32x8 z8 = {0, 0, 0, 0, 0, 0, 0, 0};
  __builtin_amdgcn_tensor_load_to_lds(g0, g1, z, z, z8, 0);
#else
  __builtin_amdgcn_tensor_load_to_lds(g0, g1, z, z, 0);
#endif
}
#endif // HAVE_TDM

// ---------- kernels ----------

__global__ void embed_kernel(const int* __restrict__ tok, const float* __restrict__ emb,
                             float* __restrict__ X) {
  int row = blockIdx.x;
  int l = row % SEQ;
  const float* e = emb + (size_t)tok[row] * HID;
  float* x = X + (size_t)row * HID;
  for (int i = threadIdx.x; i < HID; i += blockDim.x) {
    int pair = i >> 1;
    float ang = (float)l * powf(10000.0f, -(float)(2 * pair) / (float)HID);
    float pe = (i & 1) ? cosf(ang) : sinf(ang);
    x[i] = e[i] + pe;
  }
}

__global__ void layernorm_kernel(const float* __restrict__ X, const float* __restrict__ g,
                                 const float* __restrict__ bt, float* __restrict__ Y,
                                 int nrows) {
  int wave = threadIdx.x >> 5, lane = threadIdx.x & 31;
  int row = blockIdx.x * 8 + wave;
  if (row >= nrows) return;
  const float* x = X + (size_t)row * HID;
  float v[24], s = 0.f;
  #pragma unroll
  for (int j = 0; j < 24; ++j) { v[j] = x[lane + 32 * j]; s += v[j]; }
  s = wave_sum(s);
  float mean = s * (1.0f / HID);
  float vs = 0.f;
  #pragma unroll
  for (int j = 0; j < 24; ++j) { float d = v[j] - mean; vs += d * d; }
  vs = wave_sum(vs);
  float rstd = rsqrtf(vs * (1.0f / HID) + 1e-5f);
  float* y = Y + (size_t)row * HID;
  #pragma unroll
  for (int j = 0; j < 24; ++j) {
    int i = lane + 32 * j;
    y[i] = (v[j] - mean) * rstd * g[i] + bt[i];
  }
}

// LDS-staged GEMM: C[M,N] = act(A[M,K] @ W[K,N] + bias) (+residual).
// 256 threads = 8 waves; workgroup tile BM x BN, wave tile 16x16, K-step 32.
// A tile (fp32) is DMA'd into LDS by the Tensor Data Mover (wave 0) while all
// waves stage + transpose + bf16-convert the W tile; fragments convert A on read.
template<int RELU, int RESID>
__global__ __launch_bounds__(256) void gemm_tile_kernel(
    const float* __restrict__ A, const float* __restrict__ W,
    const float* __restrict__ bias, const float* __restrict__ residual,
    float* __restrict__ C, int M, int N, int K) {
  __shared__ __align__(16) float  Asf[BM][BK];  // 4 KB fp32 (TDM destination)
  __shared__ __align__(16) __bf16 Ws[BN][BK];   // 4 KB bf16, transposed [n][k]
  int tid = threadIdx.x;
  int wave = tid >> 5, lane = tid & 31, half = lane >> 4, r = lane & 15;
  int wm = (wave >> 2) * 16;            // 0 | 16
  int wn = (wave & 3) * 16;             // 0 | 16 | 32 | 48
  int tm0 = blockIdx.y * BM, tn0 = blockIdx.x * BN;

  int arow = tid >> 3, acol = (tid & 7) * 4;
  const float* aP = A + (size_t)(tm0 + arow) * K + acol;
  (void)aP;
  v8f acc = {};

  for (int k0 = 0; k0 < K; k0 += BK) {
    // ---- stage A tile (32x32 fp32) ----
#if HAVE_TDM
    if (wave == 0)
      tdm_load_tile_2d(A + (size_t)tm0 * K + k0, &Asf[0][0], BM, BK, K, M);
#else
    *(float4*)&Asf[arow][acol] = *(const float4*)(aP + k0);
#endif
    // ---- stage W tile (32x64 fp32 -> bf16, transposed into LDS [n][k]) ----
    #pragma unroll
    for (int j = 0; j < 4; ++j) {
      int idx = tid + 256 * j;
      int kk = idx >> 5, n2 = (idx & 31) * 2;
      const float2 w2 = *(const float2*)(W + (size_t)(k0 + kk) * N + tn0 + n2);
      v2bf p = pk_bf16(w2.x, w2.y);
      Ws[n2][kk] = p[0];
      Ws[n2 + 1][kk] = p[1];
      if (k0 + BK < K)
        __builtin_prefetch(W + (size_t)(k0 + BK + kk) * N + tn0 + n2, 0, 1);
    }
#if HAVE_TDM
    if (wave == 0) {
#if __has_builtin(__builtin_amdgcn_s_wait_tensorcnt)
      __builtin_amdgcn_s_wait_tensorcnt(0);
#endif
    }
#endif
    __syncthreads();
    // ---- fragments + WMMA ----
    v16bf af = load_frag_row(&Asf[wm + r][0], 0, half);  // ds_load_b128 + cvt_pk
    union { v16bf v; uint4 q[2]; } bf;
    bf.q[0] = *(const uint4*)&Ws[wn + r][half * 8];
    bf.q[1] = *(const uint4*)&Ws[wn + r][16 + half * 8];
    acc = wmma_bf16(af, bf.v, acc);
    __syncthreads();
  }

  int n = tn0 + wn + r;
  float bb = bias[n];
  #pragma unroll
  for (int e = 0; e < 8; ++e) {
    int m = tm0 + wm + e + 8 * half;
    float v = acc[e] + bb;
    if (RELU) v = fmaxf(v, 0.0f);
    if (RESID) v += residual[(size_t)m * N + n];
    C[(size_t)m * N + n] = v;
  }
}

// S[bh,q,k] = (Q . K)/sqrt(dk) with key mask (0: pad cols, 1: causal)
__global__ __launch_bounds__(32) void attn_scores_kernel(
    const float* __restrict__ Q, const float* __restrict__ Kf,
    const int* __restrict__ src, float* __restrict__ S, int masktype) {
  int lane = threadIdx.x, half = lane >> 4, r = lane & 15;
  int bh = blockIdx.z, b = bh >> 3, h = bh & 7;
  int tn = blockIdx.x, tm = blockIdx.y;
  const float* qrow = Q + ((size_t)b * SEQ + tm * 16 + r) * HID + h * DKH;
  const float* krow = Kf + ((size_t)b * SEQ + tn * 16 + r) * HID + h * DKH;
  v8f acc = {};
  #pragma unroll
  for (int k0 = 0; k0 < DKH; k0 += 32) {
    v16bf af = load_frag_row(qrow, k0, half);
    v16bf bf = load_frag_row(krow, k0, half);  // K^T column frag == row frag of K
    acc = wmma_bf16(af, bf, acc);
  }
  const float scale = 0.10206207262f; // 1/sqrt(96)
  int n = tn * 16 + r;
  bool colmask = (masktype == 0) && (src[b * SEQ + n] == PAD_TOK);
  float ninf = -__builtin_huge_valf();
  #pragma unroll
  for (int e = 0; e < 8; ++e) {
    int m = tm * 16 + e + 8 * half;
    float sv = acc[e] * scale;
    if (colmask) sv = ninf;
    if (masktype == 1 && n > m) sv = ninf;
    S[((size_t)bh * SEQ + m) * SEQ + n] = sv;
  }
}

// per-row (one wave): softmax -> top-13 (+26 random cols) * 1.2 -> renormalize
__global__ void softmax_moderate_kernel(float* __restrict__ S, const int* __restrict__ ridx) {
  int wave = threadIdx.x >> 5, lane = threadIdx.x & 31;
  int row = blockIdx.x * 8 + wave;
  float* p = S + (size_t)row * SEQ;
  float v[8];
  const float4* p4 = (const float4*)(p + lane * 8);
  float4 x0 = p4[0], x1 = p4[1];
  v[0]=x0.x; v[1]=x0.y; v[2]=x0.z; v[3]=x0.w; v[4]=x1.x; v[5]=x1.y; v[6]=x1.z; v[7]=x1.w;
  float mx = v[0];
  #pragma unroll
  for (int j = 1; j < 8; ++j) mx = fmaxf(mx, v[j]);
  mx = wave_max(mx);
  float s = 0.f;
  #pragma unroll
  for (int j = 0; j < 8; ++j) { v[j] = expf(v[j] - mx); s += v[j]; }
  s = wave_sum(s);
  float inv = 1.0f / s;
  #pragma unroll
  for (int j = 0; j < 8; ++j) v[j] *= inv;

  bool sel[8] = {false,false,false,false,false,false,false,false};
  for (int t = 0; t < KTOP; ++t) {
    float bv = -1.0f; int bi = 0x7fffffff;
    #pragma unroll
    for (int j = 0; j < 8; ++j)
      if (!sel[j] && v[j] > bv) { bv = v[j]; bi = lane * 8 + j; }
    #pragma unroll
    for (int m = 16; m > 0; m >>= 1) {       // argmax, tie -> lowest index
      float ov = __shfl_xor(bv, m, 32);
      int   oi = __shfl_xor(bi, m, 32);
      if (ov > bv || (ov == bv && oi < bi)) { bv = ov; bi = oi; }
    }
    if ((bi >> 3) == lane) sel[bi & 7] = true;
  }
  for (int t = 0; t < NRAND; ++t) {
    int c = ridx[t];
    if ((c >> 3) == lane) sel[c & 7] = true;
  }
  float s2 = 0.f;
  #pragma unroll
  for (int j = 0; j < 8; ++j) { if (sel[j]) v[j] *= MOD_SCALE; s2 += v[j]; }
  s2 = wave_sum(s2);
  inv = 1.0f / s2;
  float4 y0, y1;
  y0.x=v[0]*inv; y0.y=v[1]*inv; y0.z=v[2]*inv; y0.w=v[3]*inv;
  y1.x=v[4]*inv; y1.y=v[5]*inv; y1.z=v[6]*inv; y1.w=v[7]*inv;
  float4* q4 = (float4*)(p + lane * 8);
  q4[0] = y0; q4[1] = y1;
}

// O[b,q,h*96+d] = sum_k A[bh,q,k] * V[b,k,h*96+d]
__global__ __launch_bounds__(32) void attn_av_kernel(
    const float* __restrict__ S, const float* __restrict__ Vf, float* __restrict__ O) {
  int lane = threadIdx.x, half = lane >> 4, r = lane & 15;
  int bh = blockIdx.z, b = bh >> 3, h = bh & 7;
  int tn = blockIdx.x, tm = blockIdx.y;
  const float* arow = S + ((size_t)bh * SEQ + tm * 16 + r) * SEQ;
  const float* vcol = Vf + (size_t)b * SEQ * HID + h * DKH + tn * 16 + r;
  v8f acc = {};
  for (int k0 = 0; k0 < SEQ; k0 += 32) {
    v16bf af = load_frag_row(arow, k0, half);
    v16bf bf = load_frag_col(vcol, HID, k0, half);
    acc = wmma_bf16(af, bf, acc);
  }
  int n = tn * 16 + r;
  #pragma unroll
  for (int e = 0; e < 8; ++e) {
    int m = tm * 16 + e + 8 * half;
    O[((size_t)b * SEQ + m) * HID + h * DKH + n] = acc[e];
  }
}

__global__ void log_softmax_kernel(float* __restrict__ X) {
  __shared__ float red[256];
  int tid = threadIdx.x;
  float* x = X + (size_t)blockIdx.x * VOC;
  float mx = -3.4e38f;
  for (int i = tid; i < VOC; i += 256) mx = fmaxf(mx, x[i]);
  red[tid] = mx; __syncthreads();
  for (int s = 128; s > 0; s >>= 1) { if (tid < s) red[tid] = fmaxf(red[tid], red[tid + s]); __syncthreads(); }
  mx = red[0]; __syncthreads();
  float sum = 0.f;
  for (int i = tid; i < VOC; i += 256) sum += expf(x[i] - mx);
  red[tid] = sum; __syncthreads();
  for (int s = 128; s > 0; s >>= 1) { if (tid < s) red[tid] += red[tid + s]; __syncthreads(); }
  float lse = mx + logf(red[0]);
  for (int i = tid; i < VOC; i += 256) x[i] = x[i] - lse;
}

// ---------- host orchestration ----------

extern "C" void kernel_launch(void* const* d_in, const int* in_sizes, int n_in,
                              void* d_out, int out_size, void* d_ws, size_t ws_size,
                              hipStream_t stream) {
  (void)in_sizes; (void)n_in; (void)out_size; (void)ws_size;
  const int* src = (const int*)d_in[0];
  const int* tgt = (const int*)d_in[1];
  const int* rnd = (const int*)d_in[2];

  struct Lin { const float *w, *b; };
  struct AttnP { Lin q, k, v, o; };
  struct FfnP { Lin w1, w2; };
  struct LnP { const float *g, *b; };
  struct EncP { AttnP attn; FfnP ffn; LnP ln; };
  struct DecP { AttnP a1, a2; FfnP ffn; LnP ln; };

  // params leaves: top-level dict insertion order; nested dicts in JAX sorted-key
  // pytree order.  dec -> emb_dec -> emb_enc -> enc -> gen; attn{k,o,q,v}, lin{b,w},
  // ffn{w1,w2}, ln{b,g}; dec layer {attn1,attn2,ffn,ln}; enc layer {attn,ffn,ln}.
  int pi = 3;
  auto nxt    = [&]() { return (const float*)d_in[pi++]; };
  auto getLin = [&](Lin& L) { L.b = nxt(); L.w = nxt(); };
  auto getAttn= [&](AttnP& A) { getLin(A.k); getLin(A.o); getLin(A.q); getLin(A.v); };
  auto getFfn = [&](FfnP& F) { getLin(F.w1); getLin(F.w2); };
  auto getLn  = [&](LnP& L) { L.b = nxt(); L.g = nxt(); };

  DecP dec[NLAYERS];
  for (int i = 0; i < NLAYERS; ++i) { getAttn(dec[i].a1); getAttn(dec[i].a2); getFfn(dec[i].ffn); getLn(dec[i].ln); }
  const float* emb_dec = nxt();
  const float* emb_enc = nxt();
  EncP enc[NLAYERS];
  for (int i = 0; i < NLAYERS; ++i) { getAttn(enc[i].attn); getFfn(enc[i].ffn); getLn(enc[i].ln); }
  LnP gen_ln; getLn(gen_ln);
  Lin gen_out; getLin(gen_out);

  const size_t NTOK = (size_t)B_ * SEQ;      // 2048 rows
  const size_t ACT  = NTOK * HID;
  float* ws = (float*)d_ws;
  float* X  = ws;                                   // encoder state
  float* Y  = X + ACT;                              // decoder state
  float* HN = Y + ACT;                              // layernorm output
  float* Qb = HN + ACT;
  float* Kb = Qb + ACT;
  float* Vb = Kb + ACT;
  float* Ob = Vb + ACT;
  float* Sc = Ob + ACT;                             // scores: 64*256*256
  float* Fm = Sc + (size_t)B_ * NHEADS * SEQ * SEQ; // ffn mid: 2048*3072

  auto LNORM = [&](const float* in, const LnP& p) {
    layernorm_kernel<<<(int)(NTOK / 8), 256, 0, stream>>>(in, p.g, p.b, HN, (int)NTOK);
  };
  auto GEMM = [&](const float* A, const Lin& Wt, float* Cc, int M, int N, int K,
                  int relu, const float* res) {
    dim3 g(N / BN, M / BM);
    if (res)
      gemm_tile_kernel<0,1><<<g, 256, 0, stream>>>(A, Wt.w, Wt.b, res, Cc, M, N, K);
    else if (relu)
      gemm_tile_kernel<1,0><<<g, 256, 0, stream>>>(A, Wt.w, Wt.b, nullptr, Cc, M, N, K);
    else
      gemm_tile_kernel<0,0><<<g, 256, 0, stream>>>(A, Wt.w, Wt.b, nullptr, Cc, M, N, K);
  };
  auto ATTN = [&](const float* qx, const float* kx, const AttnP& P, float* xio,
                  int masktype, const int* ridx) {
    GEMM(qx, P.q, Qb, (int)NTOK, HID, HID, 0, nullptr);
    GEMM(kx, P.k, Kb, (int)NTOK, HID, HID, 0, nullptr);
    GEMM(kx, P.v, Vb, (int)NTOK, HID, HID, 0, nullptr);
    attn_scores_kernel<<<dim3(SEQ / 16, SEQ / 16, B_ * NHEADS), 32, 0, stream>>>(Qb, Kb, src, Sc, masktype);
    softmax_moderate_kernel<<<(B_ * NHEADS * SEQ) / 8, 256, 0, stream>>>(Sc, ridx);
    attn_av_kernel<<<dim3(DKH / 16, SEQ / 16, B_ * NHEADS), 32, 0, stream>>>(Sc, Vb, Ob);
    GEMM(Ob, P.o, xio, (int)NTOK, HID, HID, 0, xio);   // x = x + o@Wo + bo
  };

  // ---- encoder ----
  embed_kernel<<<(int)NTOK, 256, 0, stream>>>(src, emb_enc, X);
  for (int i = 0; i < NLAYERS; ++i) {
    LNORM(X, enc[i].ln);
    ATTN(HN, HN, enc[i].attn, X, 0, rnd + i * NRAND);
    LNORM(X, enc[i].ln);
    GEMM(HN, enc[i].ffn.w1, Fm, (int)NTOK, FFD, HID, 1, nullptr);
    GEMM(Fm, enc[i].ffn.w2, X, (int)NTOK, HID, FFD, 0, X);
  }

  // ---- decoder ----
  embed_kernel<<<(int)NTOK, 256, 0, stream>>>(tgt, emb_dec, Y);
  for (int i = 0; i < NLAYERS; ++i) {
    LNORM(Y, dec[i].ln);
    ATTN(HN, HN, dec[i].a1, Y, 1, rnd + (NLAYERS + 2 * i) * NRAND);      // causal
    LNORM(Y, dec[i].ln);
    ATTN(HN, X, dec[i].a2, Y, 0, rnd + (NLAYERS + 2 * i + 1) * NRAND);   // cross, pad mask
    LNORM(Y, dec[i].ln);
    GEMM(HN, dec[i].ffn.w1, Fm, (int)NTOK, FFD, HID, 1, nullptr);
    GEMM(Fm, dec[i].ffn.w2, Y, (int)NTOK, HID, FFD, 0, Y);
  }

  // ---- generator ----
  LNORM(Y, gen_ln);
  float* out = (float*)d_out;
  GEMM(HN, gen_out, out, (int)NTOK, VOC, HID, 0, nullptr);
  log_softmax_kernel<<<(int)NTOK, 256, 0, stream>>>(out);
}
